// GNNModel_62620623175815
// MI455X (gfx1250) — matrix operand order repaired and checked by reading, not compiled, not run on hardware
//
#include <hip/hip_runtime.h>
#include <hip/hip_bf16.h>

// ---------------------------------------------------------------------------
// GNN (node_proj + edge scatter-mean + 2x GATConv + out_proj) for gfx1250.
// GEMMs: v_wmma_f32_16x16x32_bf16, one wave per 16x64 C strip (4 n-tiles,
// A fragment reused 4x), compile-time K, unconditional b128 loads
// (A: 4x float4 per step; B: pre-packed bf16 fragments, 32B per load).
// Edge phase: bandwidth-bound scatter/gather with f32 global atomics.
// ---------------------------------------------------------------------------

#define NN   50000   // nodes
#define EE   800000  // edges (without self loops)
#define HID  256
#define NEG  0.2f

typedef __attribute__((ext_vector_type(16))) __bf16 v16bf;
typedef __attribute__((ext_vector_type(8)))  float  v8f;
typedef __attribute__((ext_vector_type(4)))  float  v4f;

static __device__ __forceinline__ void atomicMaxF(float* addr, float val) {
  unsigned* ua = (unsigned*)addr;
  unsigned old = *ua;
  while (__uint_as_float(old) < val) {
    unsigned assumed = old;
    old = atomicCAS(ua, assumed, __float_as_uint(val));
    if (old == assumed) break;
  }
}

// ---------------------------------------------------------------------------
// Pack B[Ksrc,256] (f32 row-major) into bf16 WMMA-fragment order, zero-padded
// to Kpad rows. Element (t, ks, lane, j) -> B[ks*32 + (lane>=16?8:0) + (j&7) + (j>=8?16:0),
// t*16 + (lane&15)].  Linear index i = (((t*(Kpad/32)+ks)*32)+lane)*16 + j.
// ---------------------------------------------------------------------------
__global__ void pack_B(const float* __restrict__ B, __bf16* __restrict__ out,
                       int Ksrc, int Kpad) {
  const int i = blockIdx.x * blockDim.x + threadIdx.x;
  const int total = Kpad * 256;
  if (i >= total) return;
  const int j    = i & 15;
  const int lane = (i >> 4) & 31;
  const int rest = i >> 9;                 // t*(Kpad/32) + ks
  const int ksteps = Kpad >> 5;
  const int ks = rest % ksteps;
  const int t  = rest / ksteps;
  const int n  = (t << 4) + (lane & 15);
  const int kb = (ks << 5) + ((lane >> 4) << 3) + (j & 7) + ((j >> 3) << 4);
  out[i] = (kb < Ksrc) ? (__bf16)B[(size_t)kb * 256 + n] : (__bf16)0.0f;
}

// ---------------------------------------------------------------------------
// C[M,256] = A[M,K](f32) @ Bpack(bf16 frags) (+bias).
// grid = (4, M/16), block = 32: one wave computes a 16x64 strip (4 n-tiles),
// reusing the A fragment across 4 WMMAs per k-step.  K compile-time.
// A frag (16-bit A 16x32): lane half 0 -> K 0-7 & 16-23; half 1 -> 8-15 & 24-31.
// C/D: VGPR j = row m0 + (lane>=16?8:0) + j, col = n0 + (lane&15).
// ---------------------------------------------------------------------------
template <int K>
__global__ void gemm_wmma_bf16(const float* __restrict__ A, int lda,
                               const __bf16* __restrict__ Bp,
                               const float* __restrict__ bias,
                               float* __restrict__ C, int ldc) {
  constexpr int KS = K >> 5;              // k-steps
  const int tq   = blockIdx.x;            // quad of n-tiles (0..3)
  const int m0   = blockIdx.y << 4;
  const int lane = threadIdx.x & 31;
  const int r    = lane & 15;
  const int hi   = lane >> 4;             // 0 or 1
  const int koffA = hi << 3;              // 0 / 8

  v8f acc0 = {}, acc1 = {}, acc2 = {}, acc3 = {};
  const float*  Ap = A + (size_t)(m0 + r) * (size_t)lda + koffA;
  // fragments of n-tile t start at ((t*KS)*32)*16 elements; lane stride 16
  const __bf16* Bb = Bp + ((size_t)(tq * 4) * KS * 32 + lane) * 16;
  constexpr size_t TSTRIDE = (size_t)KS * 32 * 16;  // elems between n-tiles

  for (int ks = 0; ks < KS; ++ks) {
    const int k0 = ks << 5;
    __builtin_prefetch(Ap + k0 + 32, 0, 0);          // global_prefetch_b8
    const v4f a0 = *(const v4f*)(Ap + k0);           // K rel: 0..3
    const v4f a1 = *(const v4f*)(Ap + k0 + 4);       // 4..7
    const v4f a2 = *(const v4f*)(Ap + k0 + 16);      // 16..19
    const v4f a3 = *(const v4f*)(Ap + k0 + 20);      // 20..23
    v16bf av;
    av[0]  = (__bf16)a0.x; av[1]  = (__bf16)a0.y;
    av[2]  = (__bf16)a0.z; av[3]  = (__bf16)a0.w;
    av[4]  = (__bf16)a1.x; av[5]  = (__bf16)a1.y;
    av[6]  = (__bf16)a1.z; av[7]  = (__bf16)a1.w;
    av[8]  = (__bf16)a2.x; av[9]  = (__bf16)a2.y;
    av[10] = (__bf16)a2.z; av[11] = (__bf16)a2.w;
    av[12] = (__bf16)a3.x; av[13] = (__bf16)a3.y;
    av[14] = (__bf16)a3.z; av[15] = (__bf16)a3.w;
    const __bf16* Bs = Bb + (size_t)k0 * 16;
    const v16bf bv0 = *(const v16bf*)(Bs);                 // 32B, coalesced
    const v16bf bv1 = *(const v16bf*)(Bs + TSTRIDE);
    const v16bf bv2 = *(const v16bf*)(Bs + 2 * TSTRIDE);
    const v16bf bv3 = *(const v16bf*)(Bs + 3 * TSTRIDE);
    acc0 = __builtin_amdgcn_wmma_f32_16x16x32_bf16(false, av, false, bv0,
                                                   (short)0, acc0, false, false);
    acc1 = __builtin_amdgcn_wmma_f32_16x16x32_bf16(false, av, false, bv1,
                                                   (short)0, acc1, false, false);
    acc2 = __builtin_amdgcn_wmma_f32_16x16x32_bf16(false, av, false, bv2,
                                                   (short)0, acc2, false, false);
    acc3 = __builtin_amdgcn_wmma_f32_16x16x32_bf16(false, av, false, bv3,
                                                   (short)0, acc3, false, false);
  }

  const int mb = m0 + (hi << 3);
  float* Crow = C + (size_t)mb * (size_t)ldc + (tq << 6) + r;
#pragma unroll
  for (int q = 0; q < 4; ++q) {
    const v8f acc = (q == 0) ? acc0 : (q == 1) ? acc1 : (q == 2) ? acc2 : acc3;
    const int n = (tq << 6) + (q << 4) + r;
    const float bb = bias ? bias[n] : 0.0f;
#pragma unroll
    for (int j = 0; j < 8; ++j)
      Crow[(size_t)j * (size_t)ldc + (q << 4)] = acc[j] + bb;
  }
}

// ---- scatter-mean of edge_attr over dst -----------------------------------
__global__ void scatter_edges(const float* __restrict__ edge_attr,
                              const int* __restrict__ dst,
                              float* __restrict__ accum, float* __restrict__ cnt) {
  const long long i = (long long)blockIdx.x * blockDim.x + threadIdx.x;
  if (i >= (long long)EE * 16) return;
  const int e = (int)(i >> 4);
  const int f = (int)(i & 15);
  const int d = dst[e];
  if ((unsigned)d >= (unsigned)NN) return;  // defensive (index dtype)
  atomicAdd(&accum[(size_t)d * 16 + f], edge_attr[(size_t)e * 16 + f]);
  if (f == 0) atomicAdd(&cnt[d], 1.0f);
}

// ---- mean + zero-pad to [N,32] --------------------------------------------
__global__ void nef_div(const float* __restrict__ accum,
                        const float* __restrict__ cnt, float* __restrict__ nef) {
  const int i = blockIdx.x * blockDim.x + threadIdx.x;
  if (i >= NN * 32) return;
  const int n = i >> 5, f = i & 31;
  nef[i] = (f < 16) ? accum[(size_t)n * 16 + f] / fmaxf(cnt[n], 1.0f) : 0.0f;
}

// ---- per-node attention scores: a[n,h] = <xh[n,h,:], att[h,:]> ------------
__global__ void att_scores(const float* __restrict__ xh,
                           const float* __restrict__ att_s,
                           const float* __restrict__ att_d,
                           float* __restrict__ a_src, float* __restrict__ a_dst) {
  const int i = blockIdx.x * blockDim.x + threadIdx.x;
  if (i >= NN * 4) return;
  const int n = i >> 2, h = i & 3;
  const float* xp = xh + (size_t)n * HID + h * 64;
  float ss = 0.0f, sd = 0.0f;
#pragma unroll 4
  for (int c = 0; c < 64; ++c) {
    const float v = xp[c];
    ss += v * att_s[h * 64 + c];
    sd += v * att_d[h * 64 + c];
  }
  a_src[i] = ss;
  a_dst[i] = sd;
}

// ---- reset per-layer accumulators -----------------------------------------
__global__ void init_layer(float* __restrict__ agg, float* __restrict__ m,
                           float* __restrict__ denom) {
  const int i = blockIdx.x * blockDim.x + threadIdx.x;
  if (i < NN * HID) agg[i] = 0.0f;
  if (i < NN * 4) { m[i] = -1e30f; denom[i] = 0.0f; }
}

// ---- segment max of leaky-relu logits over dst ----------------------------
__global__ void edge_max(const int* __restrict__ src, const int* __restrict__ dst,
                         const float* __restrict__ a_src,
                         const float* __restrict__ a_dst,
                         float* __restrict__ m) {
  const long long i = (long long)blockIdx.x * blockDim.x + threadIdx.x;
  const int Etot = EE + NN;  // self loops appended
  if (i >= (long long)Etot * 4) return;
  const int e = (int)(i >> 2), h = (int)(i & 3);
  int s, d;
  if (e < EE) { s = src[e]; d = dst[e]; } else { s = d = e - EE; }
  if ((unsigned)s >= (unsigned)NN || (unsigned)d >= (unsigned)NN) return;
  float l = a_src[s * 4 + h] + a_dst[d * 4 + h];
  l = (l > 0.0f) ? l : NEG * l;
  atomicMaxF(&m[d * 4 + h], l);
}

// ---- unnormalized softmax-weighted aggregation (block per edge) -----------
__global__ void edge_agg(const int* __restrict__ src, const int* __restrict__ dst,
                         const float* __restrict__ a_src,
                         const float* __restrict__ a_dst,
                         const float* __restrict__ m,
                         const float* __restrict__ xh,
                         float* __restrict__ agg, float* __restrict__ denom) {
  const int e = blockIdx.x;      // 0 .. EE+NN-1
  const int t = threadIdx.x;     // 0 .. 255 -> (head = t>>6, chan = t&63)
  const int h = t >> 6;
  int s, d;
  if (e < EE) { s = src[e]; d = dst[e]; } else { s = d = e - EE; }
  if ((unsigned)s >= (unsigned)NN || (unsigned)d >= (unsigned)NN) return;
  float l = a_src[s * 4 + h] + a_dst[d * 4 + h];
  l = (l > 0.0f) ? l : NEG * l;
  const float ex = __expf(l - m[d * 4 + h]);
  atomicAdd(&agg[(size_t)d * HID + t], xh[(size_t)s * HID + t] * ex);
  if ((t & 63) == 0) atomicAdd(&denom[d * 4 + h], ex);
}

// ---- normalize + bias + ELU (in place over agg) ---------------------------
__global__ void finalize(const float* __restrict__ denom,
                         const float* __restrict__ bias, float* __restrict__ agg) {
  const int i = blockIdx.x * blockDim.x + threadIdx.x;
  if (i >= NN * HID) return;
  const int n = i >> 8, h = (i >> 6) & 3;
  float v = agg[i] / denom[n * 4 + h] + bias[i & 255];
  agg[i] = (v > 0.0f) ? v : (__expf(v) - 1.0f);
}

// ---------------------------------------------------------------------------
extern "C" void kernel_launch(void* const* d_in, const int* in_sizes, int n_in,
                              void* d_out, int out_size, void* d_ws, size_t ws_size,
                              hipStream_t stream) {
  (void)in_sizes; (void)n_in; (void)out_size; (void)ws_size;
  const float* node_feats = (const float*)d_in[0];   // [N,128]
  const float* edge_attr  = (const float*)d_in[1];   // [E,16]
  const float* Wnp = (const float*)d_in[2];          // [128,256]
  const float* bnp = (const float*)d_in[3];
  const float* Wep = (const float*)d_in[4];          // [16,256]
  const float* bep = (const float*)d_in[5];
  const float* Wg1 = (const float*)d_in[6];          // [512,256]
  const float* as1 = (const float*)d_in[7];
  const float* ad1 = (const float*)d_in[8];
  const float* bg1 = (const float*)d_in[9];
  const float* Wg2 = (const float*)d_in[10];         // [256,256]
  const float* as2 = (const float*)d_in[11];
  const float* ad2 = (const float*)d_in[12];
  const float* bg2 = (const float*)d_in[13];
  const float* Wo  = (const float*)d_in[14];         // [256,256]
  const float* bo  = (const float*)d_in[15];
  const int*   ei  = (const int*)d_in[16];           // [2,E]
  const int* src = ei;
  const int* dst = ei + EE;

  // ---- workspace layout (f32 region then bf16 pack region), ~220 MB ------
  float* w     = (float*)d_ws;
  float* accum = w;                              // N*16
  float* cnt   = accum + (size_t)NN * 16;        // N
  float* nef   = cnt   + (size_t)NN;             // N*32 (zero-padded K)
  float* xcat  = nef   + (size_t)NN * 32;        // N*512
  float* xh    = xcat  + (size_t)NN * 512;       // N*256
  float* asrc  = xh    + (size_t)NN * 256;       // N*4
  float* adst  = asrc  + (size_t)NN * 4;         // N*4
  float* mmax  = adst  + (size_t)NN * 4;         // N*4
  float* denom = mmax  + (size_t)NN * 4;         // N*4
  float* agg   = denom + (size_t)NN * 4;         // N*256 (reused both layers)
  // bf16 packed weights, 64B aligned
  uintptr_t p = (uintptr_t)(agg + (size_t)NN * 256);
  p = (p + 63) & ~(uintptr_t)63;
  __bf16* pWnp = (__bf16*)p;                     // 128*256
  __bf16* pWep = pWnp + (size_t)128 * 256;       // 32*256
  __bf16* pWg1 = pWep + (size_t)32  * 256;       // 512*256
  __bf16* pWg2 = pWg1 + (size_t)512 * 256;       // 256*256
  __bf16* pWo  = pWg2 + (size_t)256 * 256;       // 256*256

  const dim3 gemm_grid(4, NN / 16);              // 4 strip-quads x 3125 rows
  const int  T = 256;
  const int  Etot = EE + NN;

  // ---- pack weights into WMMA fragment order ------------------------------
  pack_B<<<(128 * 256 + T - 1) / T, T, 0, stream>>>(Wnp, pWnp, 128, 128);
  pack_B<<<(32  * 256 + T - 1) / T, T, 0, stream>>>(Wep, pWep, 16,  32);
  pack_B<<<(512 * 256 + T - 1) / T, T, 0, stream>>>(Wg1, pWg1, 512, 512);
  pack_B<<<(256 * 256 + T - 1) / T, T, 0, stream>>>(Wg2, pWg2, 256, 256);
  pack_B<<<(256 * 256 + T - 1) / T, T, 0, stream>>>(Wo,  pWo,  256, 256);

  // ---- edge-feature scatter mean + projections into xcat ------------------
  hipMemsetAsync(accum, 0, ((size_t)NN * 16 + NN) * sizeof(float), stream);
  scatter_edges<<<((long long)EE * 16 + T - 1) / T, T, 0, stream>>>(edge_attr, dst, accum, cnt);
  nef_div<<<(NN * 32 + T - 1) / T, T, 0, stream>>>(accum, cnt, nef);
  gemm_wmma_bf16<128><<<gemm_grid, 32, 0, stream>>>(node_feats, 128, pWnp, bnp, xcat,       512);
  gemm_wmma_bf16<32> <<<gemm_grid, 32, 0, stream>>>(nef,        32,  pWep, bep, xcat + 256, 512);

  // ---- GAT layer 1 --------------------------------------------------------
  gemm_wmma_bf16<512><<<gemm_grid, 32, 0, stream>>>(xcat, 512, pWg1, nullptr, xh, 256);
  att_scores<<<(NN * 4 + T - 1) / T, T, 0, stream>>>(xh, as1, ad1, asrc, adst);
  init_layer<<<(NN * HID + T - 1) / T, T, 0, stream>>>(agg, mmax, denom);
  edge_max<<<((long long)Etot * 4 + T - 1) / T, T, 0, stream>>>(src, dst, asrc, adst, mmax);
  edge_agg<<<Etot, T, 0, stream>>>(src, dst, asrc, adst, mmax, xh, agg, denom);
  finalize<<<(NN * HID + T - 1) / T, T, 0, stream>>>(denom, bg1, agg);   // agg := y1

  // ---- GAT layer 2 --------------------------------------------------------
  gemm_wmma_bf16<256><<<gemm_grid, 32, 0, stream>>>(agg, 256, pWg2, nullptr, xh, 256);
  att_scores<<<(NN * 4 + T - 1) / T, T, 0, stream>>>(xh, as2, ad2, asrc, adst);
  init_layer<<<(NN * HID + T - 1) / T, T, 0, stream>>>(agg, mmax, denom);
  edge_max<<<((long long)Etot * 4 + T - 1) / T, T, 0, stream>>>(src, dst, asrc, adst, mmax);
  edge_agg<<<Etot, T, 0, stream>>>(src, dst, asrc, adst, mmax, xh, agg, denom);
  finalize<<<(NN * HID + T - 1) / T, T, 0, stream>>>(denom, bg2, agg);   // agg := y2

  // ---- out_proj -> d_out --------------------------------------------------
  gemm_wmma_bf16<256><<<gemm_grid, 32, 0, stream>>>(agg, 256, pWo, bo, (float*)d_out, 256);
}
